// SDKModule_8392366096852
// MI455X (gfx1250) — matrix operand order repaired and checked by reading, not compiled, not run on hardware
//
#include <hip/hip_runtime.h>

#define HWSZ 65536
#define NP   262144   // N*H*W = 4*256*256

typedef __attribute__((ext_vector_type(16))) __bf16 v16bf;
typedef __attribute__((ext_vector_type(8)))  float  v8f;

static __device__ __forceinline__ unsigned short f2bf(float f){
  unsigned u = __builtin_bit_cast(unsigned, f);
  unsigned r = 0x7FFFu + ((u >> 16) & 1u);
  return (unsigned short)((u + r) >> 16);
}
static __device__ __forceinline__ float bf2f(unsigned short h){
  unsigned u = ((unsigned)h) << 16;
  return __builtin_bit_cast(float, u);
}

// ---------------------------------------------------------------- packing ---
// ms: NCHW f32 -> NHWC bf16, LDS-transposed so both sides are coalesced.
// Block: 256 threads, tile = 64 pixels x 128 channels. LDS rows padded to 136
// halves (272 B) for bank-conflict-free access and 16 B alignment.
__global__ void k_pack_ms(const float* __restrict__ src, unsigned short* __restrict__ dst){
  __shared__ __align__(16) unsigned short sm[64 * 136];
  int tid = threadIdx.x;
  long pix0 = (long)blockIdx.x * 64;          // 4096 blocks
  int n   = (int)(pix0 >> 16);
  int hw0 = (int)(pix0 & 65535);
  // coalesced NCHW reads (lanes sweep hw at fixed c)
  for (int i = tid; i < 8192; i += 256){
    int hwo = i & 63, c = i >> 6;
    float f = src[((long)(n * 128 + c)) * HWSZ + hw0 + hwo];
    sm[hwo * 136 + c] = f2bf(f);
  }
  __syncthreads();
  // coalesced NHWC writes: 16 B per lane, contiguous across lanes
  for (int i = tid; i < 1024; i += 256){
    int hwo = i >> 4, c8 = (i & 15) * 8;
    *(uint4*)(dst + (pix0 + hwo) * 128 + c8) = *(const uint4*)&sm[hwo * 136 + c8];
  }
}

// Pack 3x3 conv weights [M][128][3][3] into WMMA A-fragment order.
// K = tap*128 + c.  Fragment lane l: M = ot*16 + (l&15),
// half v: K32 = (l>>4)*16 + v
__global__ void k_pack3x3(const float* __restrict__ w, unsigned short* __restrict__ dst,
                          int numOt, int M){
  int idx = blockIdx.x * 256 + threadIdx.x;
  int total = 36 * numOt * 512;
  if (idx >= total) return;
  int v    = idx & 15;
  int lane = (idx >> 4) & 31;
  int rest = idx >> 9;
  int ot   = rest % numOt;
  int s    = rest / numOt;
  int hi   = lane >> 4;
  int k32  = (hi << 4) + v;
  int K    = s * 32 + k32;
  int tap  = K >> 7, c = K & 127;
  int m    = ot * 16 + (lane & 15);
  float val = (m < M) ? w[(m * 128 + c) * 9 + tap] : 0.0f;
  dst[idx] = f2bf(val);
}

// Pack 1x1 refine weights [128][256] into A-fragment order (K = c2).
__global__ void k_pack1x1(const float* __restrict__ w, unsigned short* __restrict__ dst){
  int idx = blockIdx.x * 256 + threadIdx.x;           // 8 s * 8 ot * 512
  if (idx >= 8 * 8 * 512) return;
  int v    = idx & 15;
  int lane = (idx >> 4) & 31;
  int rest = idx >> 9;
  int ot   = rest & 7;
  int s    = rest >> 3;
  int hi   = lane >> 4;
  int k32  = (hi << 4) + v;
  int K    = s * 32 + k32;
  int m    = ot * 16 + (lane & 15);
  dst[idx] = f2bf(w[m * 256 + K]);
}

// ------------------------------------- async 3-row halo staging (LDS) ------
// smem layout: [3 rows][66 px][128 ch] bf16 (50688 B), zero-filled OOB.
// In-range 16 B chunks use GLOBAL_LOAD_ASYNC_TO_LDS_B128 (ASYNCcnt-tracked,
// no VGPR round-trip); OOB chunks are zeroed with plain LDS stores.
static __device__ __forceinline__ void stage3row_async(const unsigned short* __restrict__ src_b,
                                                       unsigned short* smem,
                                                       int n, int h, int w0, int tid){
  for (int r = 0; r < 3; ++r){
    int hh = h + r - 1;
    bool hok = (unsigned)hh < 256u;
    for (int i = tid; i < 66 * 16; i += 128){   // 16 x 16B chunks per pixel
      int px = i >> 4, part = i & 15;
      int ww = w0 - 1 + px;
      unsigned short* dptr = smem + (r * 66 * 16 + i) * 8;
      if (hok && (unsigned)ww < 256u){
        const unsigned short* gptr =
            src_b + ((((long)n << 16) + hh * 256 + ww) * 128 + part * 8);
        unsigned ldsoff = (unsigned)(size_t)(void*)dptr;   // low 32 bits = LDS offset
        asm volatile("global_load_async_to_lds_b128 %0, %1, off"
                     :: "v"(ldsoff), "v"(gptr) : "memory");
      } else {
        uint4 z; z.x = z.y = z.z = z.w = 0u;
        *(uint4*)dptr = z;
      }
    }
  }
  asm volatile("s_wait_asynccnt 0x0" ::: "memory");
}

// ------------------------------------------------------- dyn-kernel GEMM ---
// conv3x3 128->9 (padded to 16 rows), implicit GEMM, K=1152 (36 steps).
__global__ void k_dyn_gemm(const unsigned short* __restrict__ ms_b,
                           const unsigned short* __restrict__ packD,
                           const float* __restrict__ dyn_b,
                           float* __restrict__ dynk2){
  __shared__ __align__(32) unsigned short smem[3 * 66 * 128];
  int tid  = threadIdx.x;
  int tile = blockIdx.x;                 // 4096 tiles of 64 px
  int n    = tile >> 10;
  int rem  = tile & 1023;
  int h    = rem >> 2;
  int w0   = (rem & 3) * 64;
  stage3row_async(ms_b, smem, n, h, w0, tid);
  __syncthreads();

  int lane = tid & 31, wave = tid >> 5;
  int hi = lane >> 4, nn = lane & 15;
  int colb = 1 + wave * 16 + nn;

  v8f acc;
  #pragma unroll
  for (int i = 0; i < 8; ++i) acc[i] = 0.0f;

  for (int s = 0; s < 36; ++s){
    int tap = s >> 2, c0 = (s & 3) * 32;
    int dy = tap / 3, dx = tap % 3 - 1;
    const v16bf b = *(const v16bf*)&smem[(dy * 66 + colb + dx) * 128 + c0 + hi * 16];
    const v16bf a = *(const v16bf*)&packD[((s * 32) + lane) * 16];
    acc = __builtin_amdgcn_wmma_f32_16x16x32_bf16(false, a, false, b, (short)0, acc, false, false);
  }

  long pix = ((long)n << 16) + h * 256 + w0 + wave * 16 + nn;
  #pragma unroll
  for (int r = 0; r < 8; ++r){
    int o = hi * 8 + r;                  // 0..15, only 0..8 consumed later
    float bia = (o < 9) ? dyn_b[o] : 0.0f;
    dynk2[(long)o * NP + pix] = acc[r] + bia;
  }
}

// ------------------------------- elementwise: L1-norm, einsum, fix, BN/ReLU -
__global__ void k_dynfix(const unsigned short* __restrict__ ms_b,
                         const float* __restrict__ dynk2,
                         const float* __restrict__ fix_w,
                         const float* __restrict__ g1, const float* __restrict__ b1,
                         const float* __restrict__ m1, const float* __restrict__ v1,
                         const float* __restrict__ g2, const float* __restrict__ b2,
                         const float* __restrict__ m2, const float* __restrict__ v2,
                         unsigned short* __restrict__ cat_b){
  int t = threadIdx.x;
  int c = t & 127;
  long pix = (long)blockIdx.x * 2 + (t >> 7);
  int n = (int)(pix >> 16);
  int hw = (int)(pix & 65535);
  int h = hw >> 8, w = hw & 255;

  float k[9]; float ssum = 0.0f;
  #pragma unroll
  for (int tap = 0; tap < 9; ++tap){
    float kv = dynk2[(long)tap * NP + pix];   // same addr across lanes: broadcast
    k[tap] = kv; ssum += fabsf(kv);
  }
  float inv = 1.0f / ssum;

  float dacc = 0.0f, facc = 0.0f;
  #pragma unroll
  for (int tap = 0; tap < 9; ++tap){
    int dy = tap / 3 - 1, dx = tap % 3 - 1;
    int hh = h + dy, ww = w + dx;
    float pv = 0.0f;
    if ((unsigned)hh < 256u && (unsigned)ww < 256u){
      long p2 = ((long)n << 16) + hh * 256 + ww;
      pv = bf2f(ms_b[p2 * 128 + c]);
    }
    dacc += k[tap] * inv * pv;
    facc += fix_w[c * 9 + tap] * pv;
  }
  float i2 = g2[c] * rsqrtf(v2[c] + 1e-5f);
  float dyn = fmaxf(dacc * i2 + (b2[c] - m2[c] * i2), 0.0f);
  float i1 = g1[c] * rsqrtf(v1[c] + 1e-5f);
  float fx  = fmaxf(facc * i1 + (b1[c] - m1[c] * i1), 0.0f);
  cat_b[pix * 256 + c]       = f2bf(dyn);   // channels 0..127 = dyn_ms
  cat_b[pix * 256 + 128 + c] = f2bf(fx);    // channels 128..255 = fix_ms
}

// --------------------------------------------------- refine 1x1 (K=256) ----
__global__ void k_refine(const unsigned short* __restrict__ cat_b,
                         const unsigned short* __restrict__ packR,
                         const float* __restrict__ refine_b,
                         const float* __restrict__ coef,
                         unsigned short* __restrict__ fused_b,
                         unsigned short* __restrict__ p_b){
  __shared__ __align__(32) unsigned short smem[64 * 256];   // 32 KB (reused for p tile)
  int tid = threadIdx.x;
  long px0 = (long)blockIdx.x * 64;
  // async stage: 64 px x 256 ch bf16, 2048 x 16B chunks
  for (int i = tid; i < 2048; i += 128){
    const unsigned short* gptr = cat_b + px0 * 256 + (long)i * 8;
    unsigned ldsoff = (unsigned)(size_t)(void*)(smem + i * 8);
    asm volatile("global_load_async_to_lds_b128 %0, %1, off"
                 :: "v"(ldsoff), "v"(gptr) : "memory");
  }
  asm volatile("s_wait_asynccnt 0x0" ::: "memory");
  __syncthreads();

  int lane = tid & 31, wave = tid >> 5;
  int hi = lane >> 4, nn = lane & 15;
  int pxl = wave * 16 + nn;

  v8f acc[8];
  #pragma unroll
  for (int ot = 0; ot < 8; ++ot)
    #pragma unroll
    for (int i = 0; i < 8; ++i) acc[ot][i] = 0.0f;

  for (int s = 0; s < 8; ++s){
    const v16bf b = *(const v16bf*)&smem[pxl * 256 + s * 32 + hi * 16];
    #pragma unroll
    for (int ot = 0; ot < 8; ++ot){
      const v16bf a = *(const v16bf*)&packR[((s * 8 + ot) * 32 + lane) * 16];
      acc[ot] = __builtin_amdgcn_wmma_f32_16x16x32_bf16(false, a, false, b, (short)0, acc[ot], false, false);
    }
  }
  __syncthreads();   // all waves done reading cat tile; smem reused for p tile

  long pix = px0 + pxl;
  int n = (int)(pix >> 16);
  int hw = (int)(pix & 65535);
  #pragma unroll
  for (int ot = 0; ot < 8; ++ot){
    #pragma unroll
    for (int r = 0; r < 8; ++r){
      int o = ot * 16 + hi * 8 + r;
      float f = acc[ot][r] + refine_b[o];
      long g = ((long)(n * 128 + o)) * HWSZ + hw;
      fused_b[g] = f2bf(f);                    // fused, NCHW bf16 (lane-contiguous)
      smem[pxl * 136 + o] = f2bf(coef[g] + f); // p tile in LDS (padded rows)
    }
  }
  __syncthreads();
  // coalesced NHWC writes of p: 16 B per lane, contiguous across lanes
  for (int i = tid; i < 1024; i += 128){
    int hwo = i >> 4, c8 = (i & 15) * 8;
    *(uint4*)(p_b + (px0 + hwo) * 128 + c8) = *(const uint4*)&smem[hwo * 136 + c8];
  }
}

// ----------------------- joint 3x3 (K=1152) + sigmoid gating epilogue ------
__global__ void k_joint(const unsigned short* __restrict__ p_b,
                        const unsigned short* __restrict__ packJ,
                        const float* __restrict__ joint_b,
                        const unsigned short* __restrict__ fused_b,
                        const float* __restrict__ coef,
                        float* __restrict__ out){
  __shared__ __align__(32) unsigned short smem[3 * 66 * 128];
  int tid  = threadIdx.x;
  int tile = blockIdx.x;
  int n    = tile >> 10;
  int rem  = tile & 1023;
  int h    = rem >> 2;
  int w0   = (rem & 3) * 64;
  // pull packed weights toward L2/L0 (global_prefetch_b8), 256 B granularity
  for (int i = tid; i < 1152; i += 128)
    __builtin_prefetch(packJ + (long)i * 128, 0, 3);
  stage3row_async(p_b, smem, n, h, w0, tid);
  __syncthreads();

  int lane = tid & 31, wave = tid >> 5;
  int hi = lane >> 4, nn = lane & 15;
  int colb = 1 + wave * 16 + nn;

  v8f acc[8];
  #pragma unroll
  for (int ot = 0; ot < 8; ++ot)
    #pragma unroll
    for (int i = 0; i < 8; ++i) acc[ot][i] = 0.0f;

  for (int s = 0; s < 36; ++s){
    int tap = s >> 2, c0 = (s & 3) * 32;
    int dy = tap / 3, dx = tap % 3 - 1;
    const v16bf b = *(const v16bf*)&smem[(dy * 66 + colb + dx) * 128 + c0 + hi * 16];
    #pragma unroll
    for (int ot = 0; ot < 8; ++ot){
      const v16bf a = *(const v16bf*)&packJ[((s * 8 + ot) * 32 + lane) * 16];
      acc[ot] = __builtin_amdgcn_wmma_f32_16x16x32_bf16(false, a, false, b, (short)0, acc[ot], false, false);
    }
  }

  int hw = h * 256 + w0 + wave * 16 + nn;
  int nbase = n * 128;
  #pragma unroll
  for (int ot = 0; ot < 8; ++ot){
    #pragma unroll
    for (int r = 0; r < 8; ++r){
      int o = ot * 16 + hi * 8 + r;
      float x = acc[ot][r] + joint_b[o];
      float p2 = 1.0f / (1.0f + __expf(-x));
      long g = ((long)(nbase + o)) * HWSZ + hw;
      float fu = bf2f(fused_b[g]);
      out[g] = (1.0f - p2) * fu + p2 * coef[g];
    }
  }
}

// ---------------------------------------------------------------- launch ---
extern "C" void kernel_launch(void* const* d_in, const int* in_sizes, int n_in,
                              void* d_out, int out_size, void* d_ws, size_t ws_size,
                              hipStream_t stream) {
  const float* coef     = (const float*)d_in[0];
  const float* ms       = (const float*)d_in[1];
  const float* fix_w    = (const float*)d_in[2];
  const float* bn1_g    = (const float*)d_in[3];
  const float* bn1_b    = (const float*)d_in[4];
  const float* bn1_m    = (const float*)d_in[5];
  const float* bn1_v    = (const float*)d_in[6];
  const float* dyn_w    = (const float*)d_in[7];
  const float* dyn_b    = (const float*)d_in[8];
  const float* bn2_g    = (const float*)d_in[9];
  const float* bn2_b    = (const float*)d_in[10];
  const float* bn2_m    = (const float*)d_in[11];
  const float* bn2_v    = (const float*)d_in[12];
  const float* refine_w = (const float*)d_in[13];
  const float* refine_b = (const float*)d_in[14];
  const float* joint_w  = (const float*)d_in[15];
  const float* joint_b  = (const float*)d_in[16];
  float* out = (float*)d_out;

  char* ws = (char*)d_ws;
  // ms_b region is reused for p_b (ms_b dead after k_dynfix, p_b born in k_refine)
  unsigned short* ms_b    = (unsigned short*)(ws);                       // 64 MB
  unsigned short* p_b     = ms_b;
  unsigned short* cat_b   = (unsigned short*)(ws + (size_t)67108864);    // 128 MB
  unsigned short* fused_b = (unsigned short*)(ws + (size_t)201326592);   // 64 MB
  float*          dynk2   = (float*)         (ws + (size_t)268435456);   // 16 MB
  unsigned short* packJ   = (unsigned short*)(ws + (size_t)285212672);   // 288 KB
  unsigned short* packR   = (unsigned short*)(ws + (size_t)285515776);   // 64 KB
  unsigned short* packD   = (unsigned short*)(ws + (size_t)285581312);   // 36 KB

  // 1) pack ms -> NHWC bf16 (LDS transpose, coalesced both sides)
  k_pack_ms<<<4096, 256, 0, stream>>>(ms, ms_b);
  // 2) pack weights into WMMA A-fragment order
  k_pack3x3<<<576, 256, 0, stream>>>(joint_w, packJ, 8, 128);
  k_pack3x3<<<72,  256, 0, stream>>>(dyn_w,   packD, 1, 9);
  k_pack1x1<<<128, 256, 0, stream>>>(refine_w, packR);
  // 3) dyn kernel GEMM (WMMA)
  k_dyn_gemm<<<4096, 128, 0, stream>>>(ms_b, packD, dyn_b, dynk2);
  // 4) elementwise: L1 norm + per-pixel 9-tap + depthwise fix + BN/ReLU
  k_dynfix<<<131072, 256, 0, stream>>>(ms_b, dynk2, fix_w,
                                       bn1_g, bn1_b, bn1_m, bn1_v,
                                       bn2_g, bn2_b, bn2_m, bn2_v, cat_b);
  // 5) refine 1x1 GEMM (WMMA) -> fused + p
  k_refine<<<4096, 128, 0, stream>>>(cat_b, packR, refine_b, coef, fused_b, p_b);
  // 6) joint 3x3 GEMM (WMMA) + sigmoid gating -> out
  k_joint<<<4096, 128, 0, stream>>>(p_b, packJ, joint_b, fused_b, coef, out);
}
